// MolecularEncoder_2826088481346
// MI455X (gfx1250) — compile-verified, matrix operand order checked by reading, not run on hardware
//
#include <hip/hip_runtime.h>
#include <cstdint>

// ---------------------------------------------------------------------------
// MI455X (gfx1250) molecular encoder.
// All large GEMMs -> V_WMMA_F32_16X16X32_F16 (f16 in, f32 accum); tiles are
// staged into LDS with GLOBAL_LOAD_ASYNC_TO_LDS_B128 (ASYNCcnt path) and read
// as fragments via ds_load_b128; f16 activations halve HBM traffic; segment
// sums via f32 global atomics. wave32 throughout.
// ---------------------------------------------------------------------------

typedef __attribute__((ext_vector_type(16))) _Float16 v16h;
typedef __attribute__((ext_vector_type(8)))  float    v8f;

static constexpr int kN    = 10000;
static constexpr int kE    = 40000;
static constexpr int kB    = 64;
static constexpr int kHid  = 512;
static constexpr int kMsgIn = 1057;   // 2*HID + EDGE_D + 1
static constexpr int kKPad  = 1088;   // msg-in padded to multiple of 32

__device__ __forceinline__ float siluf(float x) { return x / (1.0f + __expf(-x)); }
__device__ __forceinline__ float sigm(float x)  { return 1.0f / (1.0f + __expf(-x)); }

// WMMA 16-bit A/B fragment (CDNA5 ISA 7.12.2): lane l -> row r=l&15, phase
// hi=l>>4; vector halves [0..7] live at row bytes [hi*16 .. hi*16+15] and
// halves [8..15] at [32+hi*16 .. 32+hi*16+15]. So a fragment = two 16B loads.
union HFrag { v16h v; uint4 q[2]; };

// Async copy of one 64B row (32 halfs) global -> LDS. INST_OFFSET applies to
// both the LDS and the global address (ISA 08 async addressing), so a single
// LDS-offset VGPR + 64-bit address pair covers all four b128 chunks.
__device__ __forceinline__ void async_row64(unsigned lds_off, const _Float16* gsrc)
{
    asm volatile(
        "global_load_async_to_lds_b128 %0, %1, off\n\t"
        "global_load_async_to_lds_b128 %0, %1, off offset:16\n\t"
        "global_load_async_to_lds_b128 %0, %1, off offset:32\n\t"
        "global_load_async_to_lds_b128 %0, %1, off offset:48"
        :: "v"(lds_off), "v"(gsrc) : "memory");
}

__device__ __forceinline__ void wait_async0()
{
    asm volatile("s_wait_asynccnt 0" ::: "memory");
}

// ---------------------------------------------------------------------------
// Full-tile WMMA GEMM: out[M,Nout] = act(A[M,K] @ W[Nout,K]^T + bias)
// Requires K % 32 == 0, Nout % 256 == 0. Block = 256 thr = 8 waves (2x4);
// block tile 32x256, wave tile 16x64 (4 unconditional WMMAs per K-chunk).
// W tile (256x32 f16) and A tile (32x32 f16) staged in LDS, row stride 40
// halfs (80B) so the 16 fragment rows map to 16 distinct bank groups.
// ---------------------------------------------------------------------------
__global__ void wmma_gemm_full(const _Float16* __restrict__ A,
                               const _Float16* __restrict__ W,
                               const float* __restrict__ bias,
                               float* __restrict__ outF,
                               _Float16* __restrict__ outH,
                               int M, int K, int Nout, int act)
{
    __shared__ __align__(16) _Float16 shW[256 * 40];
    __shared__ __align__(16) _Float16 shA[32 * 40];

    const int tid  = threadIdx.x;
    const int lane = tid & 31;
    const int wave = tid >> 5;
    const int wrow = wave >> 2;          // 0..1
    const int wcol = wave & 3;           // 0..3
    const int r    = lane & 15;
    const int hi   = lane >> 4;

    const int m_base = blockIdx.x * 32;
    const int n_base = blockIdx.y * 256;

    // staging: thread t stages W col n_base+t; threads 0..31 stage A rows
    const _Float16* wsrc = W + (size_t)(n_base + tid) * K;
    int arow = m_base + tid;
    if (arow > M - 1) arow = M - 1;              // clamp; stores are masked
    const _Float16* asrc = A + (size_t)arow * K;

    const unsigned ldsW = (unsigned)(uintptr_t)(shW + tid * 40);
    const unsigned ldsA = (unsigned)(uintptr_t)(shA + tid * 40);

    v8f acc[4] = {};

    for (int k0 = 0; k0 < K; k0 += 32) {
        __syncthreads();                          // previous chunk consumed
        __builtin_prefetch((const char*)(wsrc + k0) + 64, 0, 1);  // next chunk
        async_row64(ldsW, wsrc + k0);             // W tile: 256 x 64B
        if (tid < 32)
            async_row64(ldsA, asrc + k0);         // A tile: 32 x 64B
        wait_async0();
        __syncthreads();

        HFrag a;
        {
            const _Float16* ap = shA + (wrow * 16 + r) * 40;
            a.q[0] = *(const uint4*)(ap + hi * 8);
            a.q[1] = *(const uint4*)(ap + 16 + hi * 8);
        }
        HFrag b[4];
#pragma unroll
        for (int j = 0; j < 4; ++j) {             // batch the fragment loads
            const _Float16* wpr = shW + (wcol * 64 + j * 16 + r) * 40;
            b[j].q[0] = *(const uint4*)(wpr + hi * 8);
            b[j].q[1] = *(const uint4*)(wpr + 16 + hi * 8);
        }
#pragma unroll
        for (int j = 0; j < 4; ++j)               // back-to-back WMMAs
            acc[j] = __builtin_amdgcn_wmma_f32_16x16x32_f16(
                false, a.v, false, b[j].v, (short)0, acc[j], false, false);
    }

#pragma unroll
    for (int j = 0; j < 4; ++j) {
        const int n = n_base + wcol * 64 + j * 16 + r;
        const float bv = bias ? bias[n] : 0.0f;
#pragma unroll
        for (int v = 0; v < 8; ++v) {
            const int m = m_base + wrow * 16 + v + hi * 8;
            if (m < M) {
                float val = acc[j][v] + bv;
                if (act) val = siluf(val);
                if (outF) outF[(size_t)m * Nout + n] = val;
                if (outH) outH[(size_t)m * Nout + n] = (_Float16)val;
            }
        }
    }
}

// ---------------------------------------------------------------------------
// Small-N WMMA GEMM for the edge embedding: Nout == 32, K % 32 == 0.
// 8 waves stacked along M (block tile 128 rows); wave tile 16x32,
// 2 unconditional WMMAs. No barriers -> early wave return is safe.
// ---------------------------------------------------------------------------
__global__ void wmma_gemm_n32(const _Float16* __restrict__ A,
                              const _Float16* __restrict__ W,
                              const float* __restrict__ bias,
                              _Float16* __restrict__ outH,
                              int M, int K)
{
    const int lane = threadIdx.x & 31;
    const int wave = threadIdx.x >> 5;
    const int r    = lane & 15;
    const int hi   = lane >> 4;
    const int m0   = blockIdx.x * 128 + wave * 16;
    if (m0 + 16 > M) return;

    v8f acc[2] = {};
    const _Float16* Arow = A + (size_t)(m0 + r) * K;

    for (int k0 = 0; k0 < K; k0 += 32) {
        HFrag a;
        a.q[0] = *(const uint4*)(Arow + k0 + hi * 8);
        a.q[1] = *(const uint4*)(Arow + k0 + 16 + hi * 8);
#pragma unroll
        for (int j = 0; j < 2; ++j) {
            HFrag b;
            const _Float16* Wrow = W + (size_t)(j * 16 + r) * K;
            b.q[0] = *(const uint4*)(Wrow + k0 + hi * 8);
            b.q[1] = *(const uint4*)(Wrow + k0 + 16 + hi * 8);
            acc[j] = __builtin_amdgcn_wmma_f32_16x16x32_f16(
                false, a.v, false, b.v, (short)0, acc[j], false, false);
        }
    }
#pragma unroll
    for (int j = 0; j < 2; ++j) {
        const int n = j * 16 + r;
        const float bv = bias ? bias[n] : 0.0f;
#pragma unroll
        for (int v = 0; v < 8; ++v) {
            const int m = m0 + v + hi * 8;
            outH[(size_t)m * 32 + n] = (_Float16)(acc[j][v] + bv);
        }
    }
}

// ---------------------------------------------------------------------------
// Elementwise / gather / scatter helpers
// ---------------------------------------------------------------------------
__global__ void zero_f32(float* p, size_t n)
{
    size_t i = blockIdx.x * (size_t)blockDim.x + threadIdx.x;
    if (i < n) p[i] = 0.0f;
}

__global__ void cvt_f16_pad(const float* __restrict__ src, _Float16* __restrict__ dst,
                            int rows, int scols, int dcols)
{
    size_t i = blockIdx.x * (size_t)blockDim.x + threadIdx.x;
    size_t total = (size_t)rows * dcols;
    if (i >= total) return;
    int rr = (int)(i / dcols), c = (int)(i % dcols);
    dst[i] = (c < scols) ? (_Float16)src[(size_t)rr * scols + c] : (_Float16)0.0f;
}

__global__ void layernorm_rows(const float* __restrict__ in,
                               const float* __restrict__ g, const float* __restrict__ b,
                               float* __restrict__ out32, _Float16* __restrict__ out16, int D)
{
    __shared__ float sbuf[256];
    const int row = blockIdx.x;
    const float* pr = in + (size_t)row * D;
    float s = 0.0f;
    for (int d = threadIdx.x; d < D; d += 256) s += pr[d];
    sbuf[threadIdx.x] = s; __syncthreads();
    for (int st = 128; st > 0; st >>= 1) {
        if (threadIdx.x < st) sbuf[threadIdx.x] += sbuf[threadIdx.x + st];
        __syncthreads();
    }
    const float mu = sbuf[0] / D; __syncthreads();
    float v = 0.0f;
    for (int d = threadIdx.x; d < D; d += 256) { float t = pr[d] - mu; v += t * t; }
    sbuf[threadIdx.x] = v; __syncthreads();
    for (int st = 128; st > 0; st >>= 1) {
        if (threadIdx.x < st) sbuf[threadIdx.x] += sbuf[threadIdx.x + st];
        __syncthreads();
    }
    const float rinv = rsqrtf(sbuf[0] / D + 1e-5f);
    for (int d = threadIdx.x; d < D; d += 256) {
        float o = (pr[d] - mu) * rinv * g[d] + b[d];
        out32[(size_t)row * D + d] = o;
        out16[(size_t)row * D + d] = (_Float16)o;
    }
}

// msg_in[e] = [ h[col], h[row], dist, e_feat, 0-pad ]  (f16, width kKPad)
__global__ void build_msgin(const _Float16* __restrict__ h16,
                            const _Float16* __restrict__ e16,
                            const float* __restrict__ pos,
                            const int* __restrict__ ei,
                            _Float16* __restrict__ msgin)
{
    const int e = blockIdx.x;
    const int rowi = ei[e];         // source j
    const int coli = ei[kE + e];    // target i
    float dx = pos[coli * 3 + 0] - pos[rowi * 3 + 0];
    float dy = pos[coli * 3 + 1] - pos[rowi * 3 + 1];
    float dz = pos[coli * 3 + 2] - pos[rowi * 3 + 2];
    float dist = sqrtf(dx * dx + dy * dy + dz * dz);
    dist = fminf(fmaxf(dist, 1e-4f), 100.0f);
    _Float16* dst = msgin + (size_t)e * kKPad;
    for (int d = threadIdx.x; d < kKPad; d += blockDim.x) {
        _Float16 val;
        if (d < kHid)            val = h16[(size_t)coli * kHid + d];
        else if (d < 2 * kHid)   val = h16[(size_t)rowi * kHid + (d - kHid)];
        else if (d == 2 * kHid)  val = (_Float16)dist;
        else if (d < kMsgIn)     val = e16[(size_t)e * 32 + (d - 2 * kHid - 1)];
        else                     val = (_Float16)0.0f;
        dst[d] = val;
    }
}

__global__ void scatter_add_msg(const _Float16* __restrict__ m16,
                                const int* __restrict__ ei,
                                float* __restrict__ agg)
{
    const int e = blockIdx.x;
    const int coli = ei[kE + e];
    const _Float16* src = m16 + (size_t)e * kHid;
    float* dst = agg + (size_t)coli * kHid;
    for (int d = threadIdx.x; d < kHid; d += blockDim.x)
        atomicAdd(dst + d, (float)src[d]);
}

__global__ void build_cat(const _Float16* __restrict__ h16, const float* __restrict__ agg,
                          _Float16* __restrict__ cat)
{
    size_t i = blockIdx.x * (size_t)blockDim.x + threadIdx.x;
    size_t total = (size_t)kN * 1024;
    if (i >= total) return;
    int row = (int)(i >> 10), c = (int)(i & 1023);
    cat[i] = (c < kHid) ? h16[(size_t)row * kHid + c]
                        : (_Float16)agg[(size_t)row * kHid + (c - kHid)];
}

__global__ void pos_edge_update(const _Float16* __restrict__ p1,
                                const float* __restrict__ w2, const float* __restrict__ b2,
                                const float* __restrict__ pos, const int* __restrict__ ei,
                                float* __restrict__ pacc, float* __restrict__ cnt)
{
    const int e = blockIdx.x * blockDim.x + threadIdx.x;
    if (e >= kE) return;
    const int rowi = ei[e], coli = ei[kE + e];
    float s = b2[0];
    const _Float16* pe = p1 + (size_t)e * 256;
    for (int d = 0; d < 256; ++d) s += (float)pe[d] * w2[d];
    float dx = pos[rowi * 3 + 0] - pos[coli * 3 + 0];
    float dy = pos[rowi * 3 + 1] - pos[coli * 3 + 1];
    float dz = pos[rowi * 3 + 2] - pos[coli * 3 + 2];
    float nrm = fmaxf(sqrtf(dx * dx + dy * dy + dz * dz), 1e-4f);
    float inv = s / nrm;
    float ux = fminf(fmaxf(dx * inv, -10.0f), 10.0f);
    float uy = fminf(fmaxf(dy * inv, -10.0f), 10.0f);
    float uz = fminf(fmaxf(dz * inv, -10.0f), 10.0f);
    atomicAdd(&pacc[coli * 3 + 0], ux);
    atomicAdd(&pacc[coli * 3 + 1], uy);
    atomicAdd(&pacc[coli * 3 + 2], uz);
    atomicAdd(&cnt[coli], 1.0f);
}

__global__ void pos_apply(float* __restrict__ pos, const float* __restrict__ pacc,
                          const float* __restrict__ cnt)
{
    const int i = blockIdx.x * blockDim.x + threadIdx.x;
    if (i >= kN) return;
    const float c = fmaxf(cnt[i], 1.0f);
    pos[i * 3 + 0] += pacc[i * 3 + 0] / c;
    pos[i * 3 + 1] += pacc[i * 3 + 1] / c;
    pos[i * 3 + 2] += pacc[i * 3 + 2] / c;
}

// ---------------- attention pooling ----------------
__global__ void attn_scores(const float* __restrict__ kbuf, const float* __restrict__ q,
                            const int* __restrict__ batch,
                            float* __restrict__ aexp, float* __restrict__ denom)
{
    const int i = blockIdx.x * blockDim.x + threadIdx.x;   // node*8 + head
    if (i >= kN * 8) return;
    const int n = i >> 3, h = i & 7;
    const float* kk = kbuf + (size_t)n * kHid + h * 64;
    const float* qq = q + h * 64;
    float s = 0.0f;
    for (int d = 0; d < 64; ++d) s += qq[d] * kk[d];
    s = fminf(fmaxf(s * 0.125f, -20.0f), 20.0f);           // 64^-0.5
    const float ev = __expf(s);
    aexp[i] = ev;
    atomicAdd(&denom[batch[n] * 8 + h], ev);
}

__global__ void attn_pool(const float* __restrict__ vbuf, const float* __restrict__ aexp,
                          const float* __restrict__ denom, const int* __restrict__ batch,
                          float* __restrict__ pooled)
{
    const int n = blockIdx.x;
    const int b = batch[n];
    for (int d = threadIdx.x; d < kHid; d += blockDim.x) {
        const int h = d >> 6;
        const float w = aexp[n * 8 + h] / (denom[b * 8 + h] + 1e-8f);
        atomicAdd(&pooled[(size_t)b * kHid + d], w * vbuf[(size_t)n * kHid + d]);
    }
}

// ---------------- Set2Set ----------------
__global__ void lstm_gates(const float* __restrict__ qstar, const float* __restrict__ hS,
                           const float* __restrict__ Wih, const float* __restrict__ Whh,
                           const float* __restrict__ bih, const float* __restrict__ bhh,
                           float* __restrict__ gates)
{
    const int i = blockIdx.x * blockDim.x + threadIdx.x;   // B*2048
    if (i >= kB * 2048) return;
    const int b = i >> 11, g = i & 2047;
    float s = bih[g] + bhh[g];
    const float* wi = Wih + (size_t)g * 1024;
    const float* qs = qstar + (size_t)b * 1024;
    for (int d = 0; d < 1024; ++d) s += wi[d] * qs[d];
    const float* wh = Whh + (size_t)g * kHid;
    const float* hh = hS + (size_t)b * kHid;
    for (int d = 0; d < kHid; ++d) s += wh[d] * hh[d];
    gates[i] = s;
}

__global__ void lstm_update(const float* __restrict__ gates,
                            float* __restrict__ hS, float* __restrict__ cS)
{
    const int i = blockIdx.x * blockDim.x + threadIdx.x;
    if (i >= kB * kHid) return;
    const int b = i >> 9, d = i & 511;
    const float* gg = gates + (size_t)b * 2048;
    const float ig = sigm(gg[d]);
    const float fg = sigm(gg[512 + d]);
    const float g_ = tanhf(gg[1024 + d]);
    const float og = sigm(gg[1536 + d]);
    const float c = fg * cS[i] + ig * g_;
    cS[i] = c;
    hS[i] = og * tanhf(c);
}

__global__ void s2s_scores(const float* __restrict__ h, const float* __restrict__ hS,
                           const int* __restrict__ batch,
                           float* __restrict__ aexp, float* __restrict__ denom)
{
    const int n = blockIdx.x * blockDim.x + threadIdx.x;
    if (n >= kN) return;
    const int b = batch[n];
    float s = 0.0f;
    const float* hn = h + (size_t)n * kHid;
    const float* qb = hS + (size_t)b * kHid;
    for (int d = 0; d < kHid; ++d) s += hn[d] * qb[d];
    s = fminf(fmaxf(s, -20.0f), 20.0f);
    const float ev = __expf(s);
    aexp[n] = ev;
    atomicAdd(&denom[b], ev);
}

__global__ void s2s_readout(const float* __restrict__ h, const float* __restrict__ aexp,
                            const float* __restrict__ denom, const int* __restrict__ batch,
                            float* __restrict__ rbuf)
{
    const int n = blockIdx.x;
    const int b = batch[n];
    const float w = aexp[n] / (denom[b] + 1e-8f);
    for (int d = threadIdx.x; d < kHid; d += blockDim.x)
        atomicAdd(&rbuf[(size_t)b * kHid + d], w * h[(size_t)n * kHid + d]);
}

__global__ void qstar_build(const float* __restrict__ hS, const float* __restrict__ rbuf,
                            float* __restrict__ qstar)
{
    const int i = blockIdx.x * blockDim.x + threadIdx.x;
    if (i >= kB * 1024) return;
    const int b = i >> 10, d = i & 1023;
    qstar[i] = (d < kHid) ? hS[(size_t)b * kHid + d]
                          : rbuf[(size_t)b * kHid + (d - kHid)];
}

// ---------------------------------------------------------------------------
// Host orchestration
// ---------------------------------------------------------------------------
static inline dim3 eg(size_t n) { return dim3((unsigned)((n + 255) / 256)); }

static inline void launch_gemm(const _Float16* A, const _Float16* W, const float* bias,
                               float* outF, _Float16* outH,
                               int M, int K, int Nout, int act, hipStream_t stream)
{
    dim3 grid((M + 31) / 32, Nout / 256);     // Nout % 256 == 0 for all callers
    wmma_gemm_full<<<grid, 256, 0, stream>>>(A, W, bias, outF, outH, M, K, Nout, act);
}

extern "C" void kernel_launch(void* const* d_in, const int* in_sizes, int n_in,
                              void* d_out, int out_size, void* d_ws, size_t ws_size,
                              hipStream_t stream)
{
    (void)in_sizes; (void)n_in; (void)out_size; (void)ws_size;

    // ---- inputs (setup_inputs dict order; params nested in declared order) ----
    const float* x_in      = (const float*)d_in[0];
    const float* pos_in    = (const float*)d_in[1];
    const int*   ei        = (const int*)d_in[2];
    const float* edge_attr = (const float*)d_in[3];
    const int*   batch     = (const int*)d_in[4];
    const float* atom_W = (const float*)d_in[5];
    const float* atom_b = (const float*)d_in[6];
    const float* ln_g   = (const float*)d_in[7];
    const float* ln_b   = (const float*)d_in[8];
    const float* edge_W = (const float*)d_in[9];
    const float* edge_b = (const float*)d_in[10];
    auto L = [&](int l, int k) { return (const float*)d_in[11 + 12 * l + k]; };
    const float* attn_q  = (const float*)d_in[71];
    const float* attn_kW = (const float*)d_in[72];
    const float* attn_kb = (const float*)d_in[73];
    const float* attn_vW = (const float*)d_in[74];
    const float* attn_vb = (const float*)d_in[75];
    const float* lstm_Wih = (const float*)d_in[76];
    const float* lstm_Whh = (const float*)d_in[77];
    const float* lstm_bih = (const float*)d_in[78];
    const float* lstm_bhh = (const float*)d_in[79];
    const float* local_W = (const float*)d_in[80];
    const float* local_b = (const float*)d_in[81];
    const float* glob_W  = (const float*)d_in[82];
    const float* glob_b  = (const float*)d_in[83];

    float* out = (float*)d_out;           // [64*512] z_local ++ [64*256] z_global

    // ---- workspace bump allocator ----
    char* wp = (char*)d_ws;
    auto alloc = [&](size_t bytes) -> void* {
        void* r = (void*)wp;
        wp += (bytes + 255) & ~(size_t)255;
        return r;
    };
    float*    h_pre  = (float*)alloc((size_t)kN * kHid * 4);
    float*    h32    = (float*)alloc((size_t)kN * kHid * 4);
    _Float16* h16    = (_Float16*)alloc((size_t)kN * kHid * 2);
    _Float16* x16    = (_Float16*)alloc((size_t)kN * 128 * 2);
    _Float16* ea16   = (_Float16*)alloc((size_t)kE * 32 * 2);
    _Float16* e16    = (_Float16*)alloc((size_t)kE * 32 * 2);
    float*    pos    = (float*)alloc((size_t)kN * 3 * 4);
    float*    pacc   = (float*)alloc((size_t)kN * 3 * 4);
    float*    cnt    = (float*)alloc((size_t)kN * 4);
    _Float16* msgin  = (_Float16*)alloc((size_t)kE * kKPad * 2);
    _Float16* m1_16  = (_Float16*)alloc((size_t)kE * kHid * 2);
    _Float16* m16    = (_Float16*)alloc((size_t)kE * kHid * 2);
    float*    agg    = (float*)alloc((size_t)kN * kHid * 4);
    _Float16* cat16  = (_Float16*)alloc((size_t)kN * 1024 * 2);
    _Float16* u116   = (_Float16*)alloc((size_t)kN * kHid * 2);
    _Float16* p116   = (_Float16*)alloc((size_t)kE * 256 * 2);
    _Float16* watom  = (_Float16*)alloc((size_t)kHid * 128 * 2);
    _Float16* wedge  = (_Float16*)alloc((size_t)32 * 32 * 2);
    _Float16* w1     = (_Float16*)alloc((size_t)kHid * kKPad * 2);
    _Float16* w2     = (_Float16*)alloc((size_t)kHid * kHid * 2);
    _Float16* wu1    = (_Float16*)alloc((size_t)kHid * 1024 * 2);
    _Float16* wu2    = (_Float16*)alloc((size_t)kHid * kHid * 2);
    _Float16* wp1    = (_Float16*)alloc((size_t)256 * kHid * 2);
    float*    kbuf   = (float*)alloc((size_t)kN * kHid * 4);
    float*    vbuf   = (float*)alloc((size_t)kN * kHid * 4);
    _Float16* wk16   = (_Float16*)alloc((size_t)kHid * kHid * 2);
    _Float16* wv16   = (_Float16*)alloc((size_t)kHid * kHid * 2);
    float*    aexp   = (float*)alloc((size_t)kN * 8 * 4);
    float*    denom  = (float*)alloc((size_t)kB * 8 * 4);
    float*    pooled = (float*)alloc((size_t)kB * kHid * 4);
    _Float16* pooled16 = (_Float16*)alloc((size_t)kB * kHid * 2);
    _Float16* wloc16 = (_Float16*)alloc((size_t)kHid * kHid * 2);
    float*    hS     = (float*)alloc((size_t)kB * kHid * 4);
    float*    cS     = (float*)alloc((size_t)kB * kHid * 4);
    float*    qstar  = (float*)alloc((size_t)kB * 1024 * 4);
    float*    gates  = (float*)alloc((size_t)kB * 2048 * 4);
    float*    aexp2  = (float*)alloc((size_t)kN * 4);
    float*    denom2 = (float*)alloc((size_t)kB * 4);
    float*    rbuf   = (float*)alloc((size_t)kB * kHid * 4);
    _Float16* qstar16 = (_Float16*)alloc((size_t)kB * 1024 * 2);
    _Float16* wglob16 = (_Float16*)alloc((size_t)256 * 1024 * 2);

    // ---- node / edge embeddings ----
    cvt_f16_pad<<<eg((size_t)kN * 128), 256, 0, stream>>>(x_in, x16, kN, 128, 128);
    cvt_f16_pad<<<eg((size_t)kHid * 128), 256, 0, stream>>>(atom_W, watom, kHid, 128, 128);
    launch_gemm(x16, watom, atom_b, h_pre, nullptr, kN, 128, kHid, 0, stream);
    layernorm_rows<<<kN, 256, 0, stream>>>(h_pre, ln_g, ln_b, h32, h16, kHid);

    cvt_f16_pad<<<eg((size_t)kE * 32), 256, 0, stream>>>(edge_attr, ea16, kE, 32, 32);
    cvt_f16_pad<<<eg(32 * 32), 256, 0, stream>>>(edge_W, wedge, 32, 32, 32);
    wmma_gemm_n32<<<dim3((kE + 127) / 128), 256, 0, stream>>>(ea16, wedge, edge_b, e16, kE, 32);

    hipMemcpyAsync(pos, pos_in, (size_t)kN * 3 * sizeof(float),
                   hipMemcpyDeviceToDevice, stream);

    // ---- 5 message-passing layers ----
    for (int l = 0; l < 5; ++l) {
        cvt_f16_pad<<<eg((size_t)kHid * kKPad), 256, 0, stream>>>(L(l, 0), w1, kHid, kMsgIn, kKPad);
        cvt_f16_pad<<<eg((size_t)kHid * kHid), 256, 0, stream>>>(L(l, 2), w2, kHid, kHid, kHid);
        cvt_f16_pad<<<eg((size_t)kHid * 1024), 256, 0, stream>>>(L(l, 4), wu1, kHid, 1024, 1024);
        cvt_f16_pad<<<eg((size_t)kHid * kHid), 256, 0, stream>>>(L(l, 6), wu2, kHid, kHid, kHid);
        cvt_f16_pad<<<eg((size_t)256 * kHid), 256, 0, stream>>>(L(l, 8), wp1, 256, kHid, kHid);

        build_msgin<<<kE, 256, 0, stream>>>(h16, e16, pos, ei, msgin);
        launch_gemm(msgin, w1, L(l, 1), nullptr, m1_16, kE, kKPad, kHid, 1, stream);
        launch_gemm(m1_16, w2, L(l, 3), nullptr, m16, kE, kHid, kHid, 1, stream);

        zero_f32<<<eg((size_t)kN * kHid), 256, 0, stream>>>(agg, (size_t)kN * kHid);
        scatter_add_msg<<<kE, 256, 0, stream>>>(m16, ei, agg);

        build_cat<<<eg((size_t)kN * 1024), 256, 0, stream>>>(h16, agg, cat16);
        launch_gemm(cat16, wu1, L(l, 5), nullptr, u116, kN, 1024, kHid, 1, stream);
        launch_gemm(u116, wu2, L(l, 7), h32, h16, kN, kHid, kHid, 0, stream);

        launch_gemm(m16, wp1, L(l, 9), nullptr, p116, kE, kHid, 256, 1, stream);
        zero_f32<<<eg((size_t)kN * 3), 256, 0, stream>>>(pacc, (size_t)kN * 3);
        zero_f32<<<eg(kN), 256, 0, stream>>>(cnt, kN);
        pos_edge_update<<<eg(kE), 256, 0, stream>>>(p116, L(l, 10), L(l, 11), pos, ei, pacc, cnt);
        pos_apply<<<eg(kN), 256, 0, stream>>>(pos, pacc, cnt);
    }

    // ---- attention pooling -> z_local ----
    cvt_f16_pad<<<eg((size_t)kHid * kHid), 256, 0, stream>>>(attn_kW, wk16, kHid, kHid, kHid);
    cvt_f16_pad<<<eg((size_t)kHid * kHid), 256, 0, stream>>>(attn_vW, wv16, kHid, kHid, kHid);
    launch_gemm(h16, wk16, attn_kb, kbuf, nullptr, kN, kHid, kHid, 0, stream);
    launch_gemm(h16, wv16, attn_vb, vbuf, nullptr, kN, kHid, kHid, 0, stream);
    zero_f32<<<eg(kB * 8), 256, 0, stream>>>(denom, kB * 8);
    attn_scores<<<eg((size_t)kN * 8), 256, 0, stream>>>(kbuf, attn_q, batch, aexp, denom);
    zero_f32<<<eg((size_t)kB * kHid), 256, 0, stream>>>(pooled, (size_t)kB * kHid);
    attn_pool<<<kN, 256, 0, stream>>>(vbuf, aexp, denom, batch, pooled);
    cvt_f16_pad<<<eg((size_t)kB * kHid), 256, 0, stream>>>(pooled, pooled16, kB, kHid, kHid);
    cvt_f16_pad<<<eg((size_t)kHid * kHid), 256, 0, stream>>>(local_W, wloc16, kHid, kHid, kHid);
    launch_gemm(pooled16, wloc16, local_b, out, nullptr, kB, kHid, kHid, 0, stream);

    // ---- Set2Set -> z_global ----
    zero_f32<<<eg((size_t)kB * kHid), 256, 0, stream>>>(hS, (size_t)kB * kHid);
    zero_f32<<<eg((size_t)kB * kHid), 256, 0, stream>>>(cS, (size_t)kB * kHid);
    zero_f32<<<eg((size_t)kB * 1024), 256, 0, stream>>>(qstar, (size_t)kB * 1024);
    for (int step = 0; step < 3; ++step) {
        lstm_gates<<<eg((size_t)kB * 2048), 256, 0, stream>>>(qstar, hS, lstm_Wih, lstm_Whh,
                                                              lstm_bih, lstm_bhh, gates);
        lstm_update<<<eg((size_t)kB * kHid), 256, 0, stream>>>(gates, hS, cS);
        zero_f32<<<eg(kB), 256, 0, stream>>>(denom2, kB);
        s2s_scores<<<eg(kN), 256, 0, stream>>>(h32, hS, batch, aexp2, denom2);
        zero_f32<<<eg((size_t)kB * kHid), 256, 0, stream>>>(rbuf, (size_t)kB * kHid);
        s2s_readout<<<kN, 256, 0, stream>>>(h32, aexp2, denom2, batch, rbuf);
        qstar_build<<<eg((size_t)kB * 1024), 256, 0, stream>>>(hS, rbuf, qstar);
    }
    cvt_f16_pad<<<eg((size_t)kB * 1024), 256, 0, stream>>>(qstar, qstar16, kB, 1024, 1024);
    cvt_f16_pad<<<eg((size_t)256 * 1024), 256, 0, stream>>>(glob_W, wglob16, 256, 1024, 1024);
    launch_gemm(qstar16, wglob16, glob_b, out + (size_t)kB * kHid, nullptr,
                kB, 1024, 256, 0, stream);
}
